// MultiheadAttentionRoPE_1236950582031
// MI455X (gfx1250) — matrix operand order repaired and checked by reading, not compiled, not run on hardware
//
#include <hip/hip_runtime.h>
#include <stdint.h>

typedef __attribute__((ext_vector_type(16))) _Float16 v16h;
typedef __attribute__((ext_vector_type(8)))  _Float16 v8h;
typedef __attribute__((ext_vector_type(4)))  _Float16 v4h;
typedef __attribute__((ext_vector_type(4)))  float    v4f;
typedef __attribute__((ext_vector_type(8)))  float    v8f;

union V16 { v16h v; v8h h[2]; };
union F8  { v8f  v; float f[8]; };
union V16E { v16h v; _Float16 e[16]; };
union V8E  { v8h  v; _Float16 e[8]; };

#define DEVINL __device__ __forceinline__

constexpr int B_  = 2;
constexpr int LQ  = 2048;
constexpr int LKV = 4096;
constexpr int C_  = 512;
constexpr int H_  = 8;
constexpr int DH  = 64;
constexpr int BH  = B_ * H_;

DEVINL v8f wmma_f16(v16h a, v16h b, v8f c) {
  return __builtin_amdgcn_wmma_f32_16x16x32_f16(false, a, false, b, (short)0, c, false, false);
}

// ---------------- f32 -> f16 conversion (x4 vectorized) ----------------
__global__ void k_cvt_f32_f16(const float* __restrict__ in, _Float16* __restrict__ out, int n4) {
  int i = blockIdx.x * blockDim.x + threadIdx.x;
  if (i < n4) {
    v4f x = *(const v4f*)(in + (size_t)i * 4);
    v4h y;
    y.x = (_Float16)x.x; y.y = (_Float16)x.y; y.z = (_Float16)x.z; y.w = (_Float16)x.w;
    *(v4h*)(out + (size_t)i * 4) = y;
  }
}

// ---------------- WMMA GEMM core: double-buffered async-to-LDS weight staging ----------------
// blockDim = 128 (4 waves). Wave w computes rows [m0w, m0w+32) x cols [n0, n0+64) of X @ W^T.
// Fully unrolled k-loop (16 stages): buffer parity and prefetch/drain decisions are static,
// so the hot loop is branch-free. Async loads retire in order -> s_wait_asynccnt<=2 means
// the current stage's tile has landed while the next stage is still in flight.
DEVINL void wmma_gemm_32x64_staged(const _Float16* __restrict__ X, int ldx,
                                   const _Float16* __restrict__ W,
                                   int m0w, int n0, F8 acc[2][4]) {
  __shared__ __align__(16) _Float16 Wl[2][64][32];
  const int tid  = threadIdx.x;
  const int lane = tid & 31, lrow = lane & 15, hi = lane >> 4;
  const int r    = tid >> 1;           // W row within tile (0..63)
  const int cc   = (tid & 1) * 16;     // element offset within row (0 or 16)
  const uint32_t ldst0 = (uint32_t)(size_t)&Wl[0][r][cc];
  const uint32_t ldst1 = (uint32_t)(size_t)&Wl[1][r][cc];
  const _Float16* wrow = W + (size_t)(n0 + r) * C_ + cc;
  // prologue: stage k0=0 into buffer 0
  asm volatile("global_load_async_to_lds_b128 %0, %1, off"
               :: "v"(ldst0), "v"((uint64_t)(size_t)wrow) : "memory");
  asm volatile("global_load_async_to_lds_b128 %0, %1, off offset:16"
               :: "v"(ldst0), "v"((uint64_t)(size_t)wrow) : "memory");
  #pragma unroll
  for (int k0 = 0; k0 < C_; k0 += 32) {
    const int buf = (k0 >> 5) & 1;     // compile-time under full unroll
    if (k0 + 32 < C_) {                // static: prefetch next stage into the other buffer
      const uint64_t gn = (uint64_t)(size_t)(wrow + k0 + 32);
      const uint32_t ld = buf ? ldst0 : ldst1;
      asm volatile("global_load_async_to_lds_b128 %0, %1, off"
                   :: "v"(ld), "v"(gn) : "memory");
      asm volatile("global_load_async_to_lds_b128 %0, %1, off offset:16"
                   :: "v"(ld), "v"(gn) : "memory");
      asm volatile("s_wait_asynccnt 0x2" ::: "memory");  // current done, next in flight
    } else {
      asm volatile("s_wait_asynccnt 0x0" ::: "memory");
    }
    __syncthreads();
    V16 a[2];
    #pragma unroll
    for (int u = 0; u < 2; ++u) {
      const _Float16* ab = X + (size_t)(m0w + u * 16 + lrow) * ldx + k0 + hi * 16;
      a[u].h[0] = *(const v8h*)(ab);
      a[u].h[1] = *(const v8h*)(ab + 8);
    }
    v16h b[4];                         // hoist all B tiles: one dscnt wait, then 8 WMMAs
    #pragma unroll
    for (int j = 0; j < 4; ++j) b[j] = *(const v16h*)(&Wl[buf][j * 16 + lrow][hi * 16]);
    #pragma unroll
    for (int j = 0; j < 4; ++j)
      #pragma unroll
      for (int u = 0; u < 2; ++u)
        acc[u][j].v = wmma_f16(a[u].v, b[j], acc[u][j].v);
    __syncthreads();                   // all reads done before this buffer is refilled
  }
}

// ---------------- Q projection (scale 1/sqrt(Dh) folded in): Qh[BH][LQ][DH] ----------------
__global__ __launch_bounds__(128) void k_proj_q(const _Float16* __restrict__ X,
                                                const _Float16* __restrict__ Wq,
                                                const float* __restrict__ bq,
                                                _Float16* __restrict__ Qh) {
  const int m0 = blockIdx.x * 128 + (threadIdx.x >> 5) * 32;
  const int h  = blockIdx.y;
  const int n0 = h * DH;
  F8 acc[2][4] = {};
  wmma_gemm_32x64_staged(X, C_, Wq, m0, n0, acc);
  const int lane = threadIdx.x & 31, lrow = lane & 15, hi = lane >> 4;
  const int b = m0 / LQ;
  #pragma unroll
  for (int u = 0; u < 2; ++u)
    #pragma unroll
    for (int j = 0; j < 4; ++j) {
      const int dh = j * 16 + lrow;
      const float bias = bq[n0 + dh];
      #pragma unroll
      for (int g = 0; g < 8; ++g) {
        const int lq = (m0 % LQ) + u * 16 + g + hi * 8;
        Qh[((size_t)(b * H_ + h) * LQ + lq) * DH + dh] =
            (_Float16)((acc[u][j].f[g] + bias) * 0.125f);
      }
    }
}

// ---------------- K projection + RoPE: Kh[BH][LKV][DH] ----------------
__global__ __launch_bounds__(128) void k_proj_k_rope(const _Float16* __restrict__ X,
                                                     const _Float16* __restrict__ Wk,
                                                     const float* __restrict__ bk,
                                                     _Float16* __restrict__ Kh) {
  const int m0 = blockIdx.x * 128 + (threadIdx.x >> 5) * 32;
  const int h  = blockIdx.y;
  const int n0 = h * DH;
  F8 acc[2][4] = {};
  wmma_gemm_32x64_staged(X, C_, Wk, m0, n0, acc);
  const int lane = threadIdx.x & 31, lrow = lane & 15, hi = lane >> 4;
  const int b = m0 / LKV;
  #pragma unroll
  for (int u = 0; u < 2; ++u)
    #pragma unroll
    for (int j = 0; j < 4; ++j) {
      const int dh = j * 16 + lrow;
      const float bias = bk[n0 + dh];
      const float invf = __powf(10000.0f, -(float)(dh >> 1) / 32.0f);
      #pragma unroll
      for (int g = 0; g < 8; ++g) {
        const int pos = (m0 % LKV) + u * 16 + g + hi * 8;
        float x = acc[u][j].f[g] + bias;   // this lane's biased value
        float p = __shfl_xor(x, 1);        // partner (dh^1), same row
        float th = (float)pos * invf;
        float c = __cosf(th), s = __sinf(th);
        float rr = (dh & 1) ? (x * c + p * s) : (x * c - p * s);
        Kh[((size_t)(b * H_ + h) * LKV + pos) * DH + dh] = (_Float16)rr;
      }
    }
}

// ---------------- V projection, transposed: Vt[BH][DH][LKV] ----------------
__global__ __launch_bounds__(128) void k_proj_v(const _Float16* __restrict__ X,
                                                const _Float16* __restrict__ Wv,
                                                const float* __restrict__ bv,
                                                _Float16* __restrict__ Vt) {
  const int m0 = blockIdx.x * 128 + (threadIdx.x >> 5) * 32;
  const int h  = blockIdx.y;
  const int n0 = h * DH;
  F8 acc[2][4] = {};
  wmma_gemm_32x64_staged(X, C_, Wv, m0, n0, acc);
  const int lane = threadIdx.x & 31, lrow = lane & 15, hi = lane >> 4;
  const int b = m0 / LKV;
  #pragma unroll
  for (int u = 0; u < 2; ++u)
    #pragma unroll
    for (int j = 0; j < 4; ++j) {
      const int dh = j * 16 + lrow;
      const float bias = bv[n0 + dh];
      #pragma unroll
      for (int g = 0; g < 8; ++g) {
        const int lkv = (m0 % LKV) + u * 16 + g + hi * 8;
        Vt[((size_t)(b * H_ + h) * DH + dh) * LKV + lkv] = (_Float16)(acc[u][j].f[g] + bias);
      }
    }
}

// ---------------- Flash attention, transposed-S design ----------------
// One wave per (bh, 32 q rows). S^T = K*Q^T puts q on lanes, kv on VGPRs: softmax stats are
// per-lane scalars (in-lane reductions + one shfl_xor(16)). O^T = V^T*P^T with P^T built in
// registers via one half-exchange shuffle. No LDS; contiguous 16B output stores.
__global__ __launch_bounds__(32) void k_attn(const _Float16* __restrict__ Qh,
                                             const _Float16* __restrict__ Kh,
                                             const _Float16* __restrict__ Vt,
                                             _Float16* __restrict__ O) { // [B][LQ][C]
  const int lane = threadIdx.x & 31, lq16 = lane & 15, hi = lane >> 4;
  const int nqb = LQ / 32;
  const int qb = blockIdx.x % nqb;
  const int bh = blockIdx.x / nqb;
  const int m0 = qb * 32;
  const _Float16* Qb = Qh + ((size_t)bh * LQ + m0) * DH;
  const _Float16* Kb = Kh + (size_t)bh * LKV * DH;
  const _Float16* Vb = Vt + (size_t)bh * DH * LKV;

  // Q^T B-operand tiles: [w = q subtile][t = dh chunk]; lane holds 32B of its q row
  v16h qB[2][2];
  #pragma unroll
  for (int w = 0; w < 2; ++w)
    #pragma unroll
    for (int t = 0; t < 2; ++t)
      qB[w][t] = *(const v16h*)(Qb + (size_t)(w * 16 + lq16) * DH + t * 32 + hi * 16);

  float m_[2] = {-3.0e38f, -3.0e38f};
  float l_[2] = {0.0f, 0.0f};
  F8 ot[2][4] = {};                    // [w][dh tile] O^T accumulators

  #pragma unroll 2
  for (int kc = 0; kc < LKV; kc += 32) {
    if (kc + 32 < LKV) {
      __builtin_prefetch(Kb + (size_t)(kc + 32 + lq16) * DH, 0, 3);
      __builtin_prefetch(Vb + (size_t)lq16 * LKV + kc + 32, 0, 3);
    }
    // ---- S^T tiles: A = K rows (16kv x 32dh), B = Q^T ----
    V16 ka[2][2];                      // [kv subtile][dh chunk]
    #pragma unroll
    for (int u2 = 0; u2 < 2; ++u2)
      #pragma unroll
      for (int t = 0; t < 2; ++t) {
        const _Float16* ab = Kb + (size_t)(kc + u2 * 16 + lq16) * DH + t * 32 + hi * 16;
        ka[u2][t].h[0] = *(const v8h*)(ab);
        ka[u2][t].h[1] = *(const v8h*)(ab + 8);
      }
    F8 st[2][2] = {};                  // [w][kv subtile]; lanes=q, VGPRs=kv
    #pragma unroll
    for (int w = 0; w < 2; ++w)
      #pragma unroll
      for (int u2 = 0; u2 < 2; ++u2)
        #pragma unroll
        for (int t = 0; t < 2; ++t)
          st[w][u2].v = wmma_f16(ka[u2][t].v, qB[w][t], st[w][u2].v);

    // ---- online softmax (per-lane scalars; scale folded into Q) + P^T build ----
    v16h pB[2];
    #pragma unroll
    for (int w = 0; w < 2; ++w) {
      float mx = -3.0e38f;
      #pragma unroll
      for (int g = 0; g < 8; ++g) mx = fmaxf(mx, fmaxf(st[w][0].f[g], st[w][1].f[g]));
      mx = fmaxf(mx, __shfl_xor(mx, 16));          // partner holds the other 16 kv
      const float mn = fmaxf(m_[w], mx);
      F8 p0, p1;
      float rs = 0.0f;
      #pragma unroll
      for (int g = 0; g < 8; ++g) { p0.f[g] = __expf(st[w][0].f[g] - mn); rs += p0.f[g]; }
      #pragma unroll
      for (int g = 0; g < 8; ++g) { p1.f[g] = __expf(st[w][1].f[g] - mn); rs += p1.f[g]; }
      rs += __shfl_xor(rs, 16);
      const float alpha = __expf(m_[w] - mn);
      l_[w] = l_[w] * alpha + rs;
      m_[w] = mn;
      #pragma unroll
      for (int j = 0; j < 4; ++j) ot[w][j].v *= alpha;   // per-lane scalar splat
      // Build B-operand P^T (32kv x 16q): one half-exchange supplies the missing 8 rows.
      float ex[8];
      #pragma unroll
      for (int g = 0; g < 8; ++g) {
        float src = hi ? p0.f[g] : p1.f[g];
        ex[g] = __shfl_xor(src, 16);
      }
      V16E pb;
      #pragma unroll
      for (int g = 0; g < 8; ++g) pb.e[g]     = (_Float16)(hi ? ex[g]     : p0.f[g]);
      #pragma unroll
      for (int g = 0; g < 8; ++g) pb.e[g + 8] = (_Float16)(hi ? p1.f[g]  : ex[g]);
      pB[w] = pb.v;
    }

    // ---- O^T += V^T * P^T; V A-tiles shared by both q subtiles ----
    #pragma unroll
    for (int j = 0; j < 4; ++j) {
      V16 va;
      const _Float16* vb = Vb + (size_t)(j * 16 + lq16) * LKV + kc + hi * 16;
      va.h[0] = *(const v8h*)(vb);
      va.h[1] = *(const v8h*)(vb + 8);
      #pragma unroll
      for (int w = 0; w < 2; ++w) ot[w][j].v = wmma_f16(va.v, pB[w], ot[w][j].v);
    }
  }

  // ---- normalize + store: O^T C-layout -> contiguous 16B stores per (w,j) ----
  const int b = bh / H_, h = bh % H_;
  #pragma unroll
  for (int w = 0; w < 2; ++w) {
    const size_t row = (size_t)(b * LQ) + m0 + w * 16 + lq16;  // this lane's q row
    const float invl = 1.0f / l_[w];
    #pragma unroll
    for (int j = 0; j < 4; ++j) {
      V8E oo;
      #pragma unroll
      for (int g = 0; g < 8; ++g) oo.e[g] = (_Float16)(ot[w][j].f[g] * invl);
      *(v8h*)(&O[row * C_ + h * DH + j * 16 + hi * 8]) = oo.v;
    }
  }
}

// ---------------- Output projection: out = AO @ Wo^T + bo (f32) ----------------
__global__ __launch_bounds__(128) void k_proj_o(const _Float16* __restrict__ X,
                                                const _Float16* __restrict__ Wo,
                                                const float* __restrict__ bo,
                                                float* __restrict__ out) {
  const int m0 = blockIdx.x * 128 + (threadIdx.x >> 5) * 32;
  const int n0 = blockIdx.y * 64;
  F8 acc[2][4] = {};
  wmma_gemm_32x64_staged(X, C_, Wo, m0, n0, acc);
  const int lane = threadIdx.x & 31, lrow = lane & 15, hi = lane >> 4;
  #pragma unroll
  for (int u = 0; u < 2; ++u)
    #pragma unroll
    for (int j = 0; j < 4; ++j) {
      const int col = n0 + j * 16 + lrow;
      const float bias = bo[col];
      #pragma unroll
      for (int g = 0; g < 8; ++g) {
        out[(size_t)(m0 + u * 16 + g + hi * 8) * C_ + col] = acc[u][j].f[g] + bias;
      }
    }
}

extern "C" void kernel_launch(void* const* d_in, const int* in_sizes, int n_in,
                              void* d_out, int out_size, void* d_ws, size_t ws_size,
                              hipStream_t stream) {
  (void)in_sizes; (void)n_in; (void)out_size; (void)ws_size;
  const float* q  = (const float*)d_in[0];
  const float* kv = (const float*)d_in[1];
  const float* Wq = (const float*)d_in[2];
  const float* bq = (const float*)d_in[3];
  const float* Wk = (const float*)d_in[4];
  const float* bk = (const float*)d_in[5];
  const float* Wv = (const float*)d_in[6];
  const float* bv = (const float*)d_in[7];
  const float* Wo = (const float*)d_in[8];
  const float* bo = (const float*)d_in[9];
  float* out = (float*)d_out;

  char* ws = (char*)d_ws;
  size_t off = 0;
  auto alloc = [&](size_t bytes) {
    char* p = ws + off;
    off += (bytes + 255) & ~(size_t)255;
    return p;
  };
  _Float16* q16  = (_Float16*)alloc((size_t)B_ * LQ  * C_ * 2);
  _Float16* kv16 = (_Float16*)alloc((size_t)B_ * LKV * C_ * 2);
  _Float16* Wq16 = (_Float16*)alloc((size_t)C_ * C_ * 2);
  _Float16* Wk16 = (_Float16*)alloc((size_t)C_ * C_ * 2);
  _Float16* Wv16 = (_Float16*)alloc((size_t)C_ * C_ * 2);
  _Float16* Wo16 = (_Float16*)alloc((size_t)C_ * C_ * 2);
  _Float16* Qh   = (_Float16*)alloc((size_t)BH * LQ  * DH * 2);
  _Float16* Kh   = (_Float16*)alloc((size_t)BH * LKV * DH * 2);
  _Float16* Vt   = (_Float16*)alloc((size_t)BH * DH * LKV * 2);
  _Float16* AO   = (_Float16*)alloc((size_t)B_ * LQ * C_ * 2);

  auto cvt = [&](const float* src, _Float16* dst, int n) {
    int n4 = n / 4;
    k_cvt_f32_f16<<<(n4 + 255) / 256, 256, 0, stream>>>(src, dst, n4);
  };
  cvt(q,  q16,  B_ * LQ  * C_);
  cvt(kv, kv16, B_ * LKV * C_);
  cvt(Wq, Wq16, C_ * C_);
  cvt(Wk, Wk16, C_ * C_);
  cvt(Wv, Wv16, C_ * C_);
  cvt(Wo, Wo16, C_ * C_);

  k_proj_q     <<<dim3(B_ * LQ  / 128, H_), 128, 0, stream>>>(q16,  Wq16, bq, Qh);
  k_proj_k_rope<<<dim3(B_ * LKV / 128, H_), 128, 0, stream>>>(kv16, Wk16, bk, Kh);
  k_proj_v     <<<dim3(B_ * LKV / 128, H_), 128, 0, stream>>>(kv16, Wv16, bv, Vt);
  k_attn       <<<dim3(BH * (LQ / 32)),     32,  0, stream>>>(Qh, Kh, Vt, AO);
  k_proj_o     <<<dim3(B_ * LQ / 128, C_ / 64), 128, 0, stream>>>(AO, Wo16, bo, out);
}
